// SpatialConv_12970801234200
// MI455X (gfx1250) — compile-verified, matrix operand order checked by reading, not run on hardware
//
#include <hip/hip_runtime.h>

// ---------------------------------------------------------------------------
// SpatialConv scan for MI455X (gfx1250): 442 sequential row/col steps, each a
// (128 x N x 1152) GEMM done with v_wmma_f32_16x16x32_bf16 (fp32 accumulate).
// ---------------------------------------------------------------------------

typedef __attribute__((ext_vector_type(16))) __bf16 v16bf;
typedef __attribute__((ext_vector_type(8)))  float  v8f;

#define Bb   32
#define Cc   128
#define Hh   64
#define Ww   160
#define KK   9
#define KDIM (Cc * KK)     // 1152 im2col K
#define NKC  (KDIM / 32)   // 36 k-chunks of 32
#define MT   (Cc / 16)     // 8 M tiles
#define PACK_PER (MT * NKC * 512)   // halfs per packed weight tensor (147456)

// f32 -> bf16 round-to-nearest-even (bit pattern)
__device__ __forceinline__ unsigned short f2bf(float f) {
    unsigned int u = __float_as_uint(f);
    u += 0x7fffu + ((u >> 16) & 1u);
    return (unsigned short)(u >> 16);
}

// ISA 16-bit A/B fragment K index for (lane, halfIdx j):
//  lane<16 : j=0..7 -> K=0..7,  j=8..15 -> K=16..23
//  lane>=16: j=0..7 -> K=8..15, j=8..15 -> K=24..31
__device__ __forceinline__ int kmap(int lane, int j) {
    return (j & 7) + ((lane >> 4) << 3) + ((j >> 3) << 4);
}

// Pack 4 weight tensors (O=128, I=128, K=9) fp32 -> bf16 in WMMA A-fragment
// order: [tensor][mtile][kchunk][lane][j] contiguous.
__global__ __launch_bounds__(256) void pack_weights_kernel(
    const float* __restrict__ wd, const float* __restrict__ wu,
    const float* __restrict__ wr, const float* __restrict__ wl,
    unsigned short* __restrict__ packed) {
    int g = blockIdx.x * blockDim.x + threadIdx.x;
    if (g >= 4 * PACK_PER) return;
    int t  = g / PACK_PER;
    int r  = g - t * PACK_PER;
    int ot = r / (NKC * 512);
    int r2 = r - ot * (NKC * 512);
    int kc   = r2 >> 9;
    int lane = (r2 >> 4) & 31;
    int j    = r2 & 15;
    int kappa = kc * 32 + kmap(lane, j);
    int o     = ot * 16 + (lane & 15);
    const float* src = (t == 0) ? wd : (t == 1) ? wu : (t == 2) ? wr : wl;
    packed[g] = f2bf(src[o * KDIM + kappa]);
}

// One scan step: dst slab += relu(conv(src slab) + bias), conv along W
// (ALONG_W=true, scanning over H) or along H (ALONG_W=false, scanning over W).
// Block = 256 threads = 8 waves; each block owns one 16-wide N tile (b fixed),
// each wave owns one 16-row M tile (output channels).
template <bool ALONG_W>
__global__ __launch_bounds__(256) void scan_step_kernel(
    float* __restrict__ x, const unsigned short* __restrict__ aw,
    const float* __restrict__ bias, int srcPos, int dstPos) {
    // staged fp32 window: 128 channels x 24 positions (16 cols + 2*4 halo)
    __shared__ __align__(16) float         xs[Cc * 24];         // 12 KB
    // B fragments in WMMA layout: [kchunk][lane][j] bf16
    __shared__ __align__(32) unsigned short lb[NKC * 512];      // 36 KB

    const int tid = threadIdx.x;
    const int n0  = blockIdx.x * 16;
    int bIdx, p0;
    if (ALONG_W) { bIdx = n0 / Ww; p0 = n0 - bIdx * Ww; }
    else         { bIdx = n0 / Hh; p0 = n0 - bIdx * Hh; }
    const int L = ALONG_W ? Ww : Hh;

    // ---- stage 1: coalesced load of src slab window into LDS (zero-padded)
    for (int e = tid; e < Cc * 24; e += 256) {
        int ic = e / 24;
        int pp = e - ic * 24;
        int p  = p0 + pp - 4;
        float v = 0.f;
        if (p >= 0 && p < L) {
            if (ALONG_W)
                v = x[((size_t)(bIdx * Cc + ic) * Hh + srcPos) * Ww + p];
            else
                v = x[((size_t)(bIdx * Cc + ic) * Hh + p) * Ww + srcPos];
        }
        xs[e] = v;
    }
    __syncthreads();

    // ---- stage 2: build all 36 B fragments (im2col, f32->bf16) in LDS
    for (int e = tid; e < NKC * 512; e += 256) {
        int kc   = e >> 9;
        int lane = (e >> 4) & 31;
        int j    = e & 15;
        int kappa = kc * 32 + kmap(lane, j);
        int ic = kappa / KK;
        int k  = kappa - ic * KK;
        int col = lane & 15;
        lb[e] = f2bf(xs[ic * 24 + col + k]);   // col+k in [0,23] always
    }
    __syncthreads();

    // ---- stage 3: 36 WMMAs per wave, A from packed global, B from LDS
    const int wave = tid >> 5;   // M tile 0..7
    const int lane = tid & 31;
    v8f acc = {};
    const unsigned short* abase = aw + (size_t)(wave * NKC) * 512 + lane * 16;
    const unsigned short* bbase = lb + lane * 16;
    for (int kc = 0; kc < NKC; ++kc) {
        v16bf afrag = *(const v16bf*)(abase + kc * 512);
        v16bf bfrag = *(const v16bf*)(bbase + kc * 512);
        acc = __builtin_amdgcn_wmma_f32_16x16x32_bf16(
            /*neg_a=*/false, afrag, /*neg_b=*/false, bfrag,
            /*c_mod=*/(short)0, acc, /*reuse_a=*/false, /*reuse_b=*/false);
    }

    // ---- epilogue: +bias, relu, accumulate into dst slab (C/D ISA layout)
    const int col  = lane & 15;
    const int half = lane >> 4;
    const int p    = p0 + col;
#pragma unroll
    for (int r = 0; r < 8; ++r) {
        int o = wave * 16 + r + 8 * half;
        float y = acc[r] + bias[o];
        y = fmaxf(y, 0.f);
        size_t idx;
        if (ALONG_W) idx = ((size_t)(bIdx * Cc + o) * Hh + dstPos) * Ww + p;
        else         idx = ((size_t)(bIdx * Cc + o) * Hh + p) * Ww + dstPos;
        x[idx] += y;
    }
}

extern "C" void kernel_launch(void* const* d_in, const int* in_sizes, int n_in,
                              void* d_out, int out_size, void* d_ws, size_t ws_size,
                              hipStream_t stream) {
    (void)in_sizes; (void)n_in; (void)out_size; (void)ws_size;
    const float* x_in = (const float*)d_in[0];
    const float* w_d  = (const float*)d_in[1];
    const float* b_d  = (const float*)d_in[2];
    const float* w_u  = (const float*)d_in[3];
    const float* b_u  = (const float*)d_in[4];
    const float* w_r  = (const float*)d_in[5];
    const float* b_r  = (const float*)d_in[6];
    const float* w_l  = (const float*)d_in[7];
    const float* b_l  = (const float*)d_in[8];

    float* x = (float*)d_out;
    unsigned short* packed = (unsigned short*)d_ws;   // needs 4*PACK_PER*2 = 1.13 MB

    // x is updated in place inside d_out (reference carry semantics).
    hipMemcpyAsync(x, x_in, (size_t)Bb * Cc * Hh * Ww * sizeof(float),
                   hipMemcpyDeviceToDevice, stream);

    pack_weights_kernel<<<(4 * PACK_PER + 255) / 256, 256, 0, stream>>>(
        w_d, w_u, w_r, w_l, packed);

    dim3 blk(256);
    const int gW = (Bb * Ww) / 16;   // 320 blocks per row step
    const int gH = (Bb * Hh) / 16;   // 128 blocks per col step

    // down: rows 1..H-1, carry = updated row i-1
    for (int i = 1; i < Hh; ++i)
        scan_step_kernel<true><<<gW, blk, 0, stream>>>(
            x, packed + 0 * PACK_PER, b_d, i - 1, i);
    // up: rows H-2..1, carry = updated row i+1 (rows 0 and H-1 untouched)
    for (int i = Hh - 2; i >= 1; --i)
        scan_step_kernel<true><<<gW, blk, 0, stream>>>(
            x, packed + 1 * PACK_PER, b_u, i + 1, i);
    // right: cols 1..W-1
    for (int j = 1; j < Ww; ++j)
        scan_step_kernel<false><<<gH, blk, 0, stream>>>(
            x, packed + 2 * PACK_PER, b_r, j - 1, j);
    // left: cols W-2..1
    for (int j = Ww - 2; j >= 1; --j)
        scan_step_kernel<false><<<gH, blk, 0, stream>>>(
            x, packed + 3 * PACK_PER, b_l, j + 1, j);
}
